// LayerBlock_27891517620993
// MI455X (gfx1250) — matrix operand order repaired and checked by reading, not compiled, not run on hardware
//
#include <hip/hip_runtime.h>
#include <hip/hip_bf16.h>
#include <cstdint>
#include <cstddef>

// ---------------------------------------------------------------------------
// Types for CDNA5 WMMA (wave32, 16x16x32 bf16 -> f32)
// ---------------------------------------------------------------------------
typedef __bf16 bf16;
typedef __attribute__((ext_vector_type(16))) __bf16 v16bf;
typedef __attribute__((ext_vector_type(8)))  __bf16 v8bf;
typedef __attribute__((ext_vector_type(8)))  float  v8f;

#define NCOEF 49          // (L+1)^2
#define CCH   128         // channels C
#define GPTS  98          // grid points 2*(L+1)^2
#define RS    (NCOEF*CCH) // 6272 row stride of [49,128] feature blocks

enum { EP_BIAS = 1, EP_SILU = 2, EP_GATE = 4, EP_F32ACC = 8, EP_F32ST = 32 };

__device__ __forceinline__ float silu_f(float v) { return v / (1.f + __expf(-v)); }

__device__ __forceinline__ v8f wmma32(v16bf a, v16bf b, v8f c) {
  return __builtin_amdgcn_wmma_f32_16x16x32_bf16(false, a, false, b, (short)0, c,
                                                 false, false);
}

// A operand (16x32, bf16), per ISA table: lanes 0-15 hold M=lane, K 0..7 & 16..23;
// lanes 16-31 hold M=lane-16, K 8..15 & 24..31. Two 16B loads per lane.
__device__ __forceinline__ v16bf frag_a(const bf16* __restrict__ A, int lda) {
  int lane = threadIdx.x & 31;
  const bf16* p = A + (long)(lane & 15) * lda + ((lane < 16) ? 0 : 8);
  v8bf lo = *(const v8bf*)(p);
  v8bf hi = *(const v8bf*)(p + 16);
  return __builtin_shufflevector(lo, hi, 0,1,2,3,4,5,6,7,8,9,10,11,12,13,14,15);
}

// B operand (32x16, bf16), B stored as [N][K] row-major: lanes 0-15 hold N=lane,
// K 0..7 & 16..23; lanes 16-31 hold N=lane-16, K 8..15 & 24..31. Two 16B loads per lane.
__device__ __forceinline__ v16bf frag_b(const bf16* __restrict__ B, int ldb) {
  int lane = threadIdx.x & 31;
  const bf16* p = B + (long)(lane & 15) * ldb + ((lane < 16) ? 0 : 8);
  v8bf lo = *(const v8bf*)(p);
  v8bf hi = *(const v8bf*)(p + 16);
  return __builtin_shufflevector(lo, hi, 0,1,2,3,4,5,6,7,8,9,10,11,12,13,14,15);
}

// One 16x16 C tile, K-loop in steps of 32
__device__ __forceinline__ v8f mm_tile(const bf16* __restrict__ A, int lda,
                                       const bf16* __restrict__ B, int ldb, int Kpad) {
  v8f acc = {0.f,0.f,0.f,0.f,0.f,0.f,0.f,0.f};
  for (int kb = 0; kb < Kpad; kb += 32)
    acc = wmma32(frag_a(A + kb, lda), frag_b(B + kb, ldb), acc);
  return acc;
}

// ---------------------------------------------------------------------------
// Batched TN GEMM over edges: C[M,N] (+)= A[M,K] * B[N,K]^T, all bf16 operands
// from global memory, fused epilogue. M multiple of 64, N multiple of 32,
// K multiple of 32. Block = 256 threads = 8 waves, block tile 64x32.
// ---------------------------------------------------------------------------
__global__ __launch_bounds__(256)
void gemm_e_kernel(const bf16* __restrict__ A, long lda, long acol,
                   const bf16* __restrict__ B, long ldb, int K, int flags,
                   const float* __restrict__ bias, long bcol,
                   const bf16* __restrict__ gate, long gld, long gcol,
                   float* __restrict__ outf, bf16* __restrict__ outb,
                   long ldc, long ccol, float sign) {
  int wave = threadIdx.x >> 5;
  int lane = threadIdx.x & 31;
  int row0 = blockIdx.y * 64 + (wave >> 1) * 16;
  int col0 = blockIdx.x * 32 + (wave & 1) * 16;
  const bf16* Ap = A + (long)row0 * lda + acol;
  const bf16* Bp = B + (long)col0 * ldb;
  v8f c = mm_tile(Ap, (int)lda, Bp, (int)ldb, K);
  int n = lane & 15;
  int mb = (lane < 16) ? 0 : 8;
  float bv = (flags & EP_BIAS) ? bias[bcol + col0 + n] : 0.f;
#pragma unroll
  for (int r = 0; r < 8; ++r) {
    long row = row0 + mb + r;
    float v = c[r] + bv;
    if (flags & EP_SILU) v = silu_f(v);
    if (flags & EP_GATE) v *= (float)gate[row * gld + gcol + col0 + n];
    if (flags & EP_F32ACC)      outf[row * ldc + ccol + col0 + n] += sign * v;
    else if (flags & EP_F32ST)  outf[row * ldc + ccol + col0 + n] = v;
    else                        outb[row * ldc + ccol + col0 + n] = (bf16)v;
  }
}

// ---------------------------------------------------------------------------
// Per-edge rotation: xrot[e, INV_PERM[i'], c] = sum_i wigner[e,i',i] * x[idx,i,c]
// One workgroup per edge; LDS-staged bf16 WMMA (M=49->64, N=128, K=49->64).
// ---------------------------------------------------------------------------
__global__ __launch_bounds__(256)
void rotate_kernel(const float* __restrict__ wigner, const float* __restrict__ x,
                   const int* __restrict__ edge_index, int E, int which,
                   const int* __restrict__ invperm, bf16* __restrict__ xrot) {
  __shared__ __align__(16) bf16 Aw[64 * 72];
  __shared__ __align__(16) bf16 Bx[128 * 72];
  int e = blockIdx.x;
  int atom = edge_index[which * E + e];
  const float* W = wigner + (long)e * NCOEF * NCOEF;
  const float* X = x + (long)atom * RS;
  for (int i = threadIdx.x; i < 64 * 72; i += 256) {
    int r = i / 72, k = i % 72;
    Aw[i] = (r < NCOEF && k < NCOEF) ? (bf16)W[r * NCOEF + k] : (bf16)0.f;
  }
  for (int i = threadIdx.x; i < 128 * 72; i += 256) {
    int ch = i / 72, k = i % 72;
    Bx[i] = (k < NCOEF) ? (bf16)X[k * CCH + ch] : (bf16)0.f;
  }
  __syncthreads();
  int wave = threadIdx.x >> 5, lane = threadIdx.x & 31;
  int n = lane & 15, mb = (lane < 16) ? 0 : 8;
  bf16* out = xrot + (long)e * RS;
  for (int t = wave; t < 32; t += 8) {
    int mt = t >> 3, nt = t & 7;
    v8f c = mm_tile(&Aw[mt * 16 * 72], 72, &Bx[nt * 16 * 72], 72, 64);
#pragma unroll
    for (int r = 0; r < 8; ++r) {
      int row = mt * 16 + mb + r;
      if (row < NCOEF) out[(long)invperm[row] * CCH + nt * 16 + n] = (bf16)c[r];
    }
  }
}

// ---------------------------------------------------------------------------
// Per-edge grid path: g=silu(tg_perm@msg); m2=fg@g; m3=winv@m2; atomicAdd->xmsg
// ---------------------------------------------------------------------------
__global__ __launch_bounds__(256)
void grid_edge_kernel(const float* __restrict__ msg, const bf16* __restrict__ tgp,
                      const bf16* __restrict__ fg, const float* __restrict__ winv,
                      const int* __restrict__ edge_index, int E,
                      float* __restrict__ xmsg) {
  __shared__ __align__(16) bf16 Areg[112 * 136];
  __shared__ __align__(16) bf16 Breg[128 * 72];
  __shared__ __align__(16) bf16 Gb[128 * 136];
  int e = blockIdx.x;
  int tgt = edge_index[E + e];
  const float* M = msg + (long)e * RS;
  int wave = threadIdx.x >> 5, lane = threadIdx.x & 31;
  int n = lane & 15, mb = (lane < 16) ? 0 : 8;

  for (int i = threadIdx.x; i < 128 * 136; i += 256) Gb[i] = (bf16)0.f;
  for (int i = threadIdx.x; i < 112 * 136; i += 256) {
    int r = i / 136, k = i % 136;
    Areg[i] = (r < GPTS && k < NCOEF) ? tgp[r * NCOEF + k] : (bf16)0.f;
  }
  for (int i = threadIdx.x; i < 128 * 72; i += 256) {
    int ch = i / 72, k = i % 72;
    Breg[i] = (k < NCOEF) ? (bf16)M[k * CCH + ch] : (bf16)0.f;
  }
  __syncthreads();
  // g = silu(tg_perm @ msg) -> store transposed Gb[c][g]
  for (int t = wave; t < 56; t += 8) {
    int mt = t / 8, nt = t % 8;
    v8f c = mm_tile(&Areg[mt * 16 * 136], 136, &Breg[nt * 16 * 72], 72, 64);
#pragma unroll
    for (int r = 0; r < 8; ++r) {
      int g = mt * 16 + mb + r;
      Gb[(nt * 16 + n) * 136 + g] = (bf16)silu_f(c[r]);
    }
  }
  __syncthreads();
  for (int i = threadIdx.x; i < 112 * 136; i += 256) {
    int r = i / 136, k = i % 136;
    Areg[i] = (r < NCOEF && k < GPTS) ? fg[r * GPTS + k] : (bf16)0.f;
  }
  __syncthreads();
  // m2 = fg @ g -> store transposed Breg[c][i]
  for (int t = wave; t < 32; t += 8) {
    int mt = t >> 3, nt = t & 7;
    v8f c = mm_tile(&Areg[mt * 16 * 136], 136, &Gb[nt * 16 * 136], 136, 128);
#pragma unroll
    for (int r = 0; r < 8; ++r) {
      int i2 = mt * 16 + mb + r;
      Breg[(nt * 16 + n) * 72 + i2] = (bf16)c[r];
    }
  }
  __syncthreads();
  const float* WI = winv + (long)e * NCOEF * NCOEF;
  for (int i = threadIdx.x; i < 112 * 136; i += 256) {
    int r = i / 136, k = i % 136;
    Areg[i] = (r < NCOEF && k < NCOEF) ? (bf16)WI[r * NCOEF + k] : (bf16)0.f;
  }
  __syncthreads();
  // m3 = winv @ m2 ; segment-sum via atomics
  for (int t = wave; t < 32; t += 8) {
    int mt = t >> 3, nt = t & 7;
    v8f c = mm_tile(&Areg[mt * 16 * 136], 136, &Breg[nt * 16 * 72], 72, 64);
#pragma unroll
    for (int r = 0; r < 8; ++r) {
      int i2 = mt * 16 + mb + r;
      if (i2 < NCOEF)
        atomicAdd(&xmsg[(long)tgt * RS + i2 * CCH + nt * 16 + n], c[r]);
    }
  }
}

// ---------------------------------------------------------------------------
// Per-atom node path: gx=tg@x, gm=tg@xmsg, 3-layer MLP, out=fg@h3
// ---------------------------------------------------------------------------
__global__ __launch_bounds__(256)
void node_kernel(const float* __restrict__ x, const float* __restrict__ xmsg,
                 const bf16* __restrict__ tg, const bf16* __restrict__ fg,
                 const bf16* __restrict__ w1, const float* __restrict__ b1,
                 const bf16* __restrict__ w2, const float* __restrict__ b2,
                 const bf16* __restrict__ w3, const float* __restrict__ b3,
                 float* __restrict__ out) {
  __shared__ __align__(16) bf16 smem[79616];
  bf16* Areg = smem;              // 112x136
  bf16* Bx   = smem + 15232;      // 128x72
  bf16* Bm   = smem + 15232 + 9216; // 128x72
  bf16* Hb   = smem + 15232;      // 112x136 (aliases Bx/Bm after they are dead)
  bf16* Gb   = smem + 33664;      // 112x264
  bf16* Hb2  = smem + 33664;      // 112x136 (aliases Gb after h1)
  bf16* H3   = smem + 63232;      // 128x128
  int na = blockIdx.x;
  int wave = threadIdx.x >> 5, lane = threadIdx.x & 31;
  int n = lane & 15, mb = (lane < 16) ? 0 : 8;
  const float* X = x + (long)na * RS;
  const float* XM = xmsg + (long)na * RS;

  for (int i = threadIdx.x; i < 128 * 128; i += 256) H3[i] = (bf16)0.f;
  for (int i = threadIdx.x; i < 112 * 136; i += 256) {
    int r = i / 136, k = i % 136;
    Areg[i] = (r < GPTS && k < NCOEF) ? tg[r * NCOEF + k] : (bf16)0.f;
  }
  for (int i = threadIdx.x; i < 128 * 72; i += 256) {
    int ch = i / 72, k = i % 72;
    bf16 vx = (k < NCOEF) ? (bf16)X[k * CCH + ch] : (bf16)0.f;
    bf16 vm = (k < NCOEF) ? (bf16)XM[k * CCH + ch] : (bf16)0.f;
    Bx[i] = vx; Bm[i] = vm;
  }
  __syncthreads();
  // gx | gm -> Gb[g][0:128 | 128:256] (A-layout rows for next GEMM)
  for (int t = wave; t < 56; t += 8) {
    int mt = t / 8, nt = t % 8;
    v8f cx = mm_tile(&Areg[mt * 16 * 136], 136, &Bx[nt * 16 * 72], 72, 64);
    v8f cm = mm_tile(&Areg[mt * 16 * 136], 136, &Bm[nt * 16 * 72], 72, 64);
#pragma unroll
    for (int r = 0; r < 8; ++r) {
      int g = mt * 16 + mb + r;
      Gb[g * 264 + nt * 16 + n]       = (bf16)cx[r];
      Gb[g * 264 + 128 + nt * 16 + n] = (bf16)cm[r];
    }
  }
  __syncthreads();
  // h1 = silu(g @ ws1^T + b1), weights streamed straight from global bf16
  for (int t = wave; t < 56; t += 8) {
    int mt = t / 8, nt = t % 8;
    v8f c = mm_tile(&Gb[mt * 16 * 264], 264, w1 + (long)(nt * 16) * 256, 256, 256);
#pragma unroll
    for (int r = 0; r < 8; ++r) {
      int g = mt * 16 + mb + r;
      Hb[g * 136 + nt * 16 + n] = (bf16)silu_f(c[r] + b1[nt * 16 + n]);
    }
  }
  __syncthreads();
  // h2 = silu(h1 @ ws2^T + b2)
  for (int t = wave; t < 56; t += 8) {
    int mt = t / 8, nt = t % 8;
    v8f c = mm_tile(&Hb[mt * 16 * 136], 136, w2 + (long)(nt * 16) * 128, 128, 128);
#pragma unroll
    for (int r = 0; r < 8; ++r) {
      int g = mt * 16 + mb + r;
      Hb2[g * 136 + nt * 16 + n] = (bf16)silu_f(c[r] + b2[nt * 16 + n]);
    }
  }
  __syncthreads();
  // h3 = h2 @ ws3^T + b3 -> store transposed H3[c][g]
  for (int t = wave; t < 56; t += 8) {
    int mt = t / 8, nt = t % 8;
    v8f c = mm_tile(&Hb2[mt * 16 * 136], 136, w3 + (long)(nt * 16) * 128, 128, 128);
#pragma unroll
    for (int r = 0; r < 8; ++r) {
      int g = mt * 16 + mb + r;
      H3[(nt * 16 + n) * 128 + g] = (bf16)(c[r] + b3[nt * 16 + n]);
    }
  }
  __syncthreads();
  for (int i = threadIdx.x; i < 112 * 136; i += 256) {
    int r = i / 136, k = i % 136;
    Areg[i] = (r < NCOEF && k < GPTS) ? fg[r * GPTS + k] : (bf16)0.f;
  }
  __syncthreads();
  // out = fg @ h3
  for (int t = wave; t < 32; t += 8) {
    int mt = t >> 3, nt = t & 7;
    v8f c = mm_tile(&Areg[mt * 16 * 136], 136, &H3[(nt * 16) * 128], 128, 128);
#pragma unroll
    for (int r = 0; r < 8; ++r) {
      int i2 = mt * 16 + mb + r;
      if (i2 < NCOEF) out[(long)na * RS + i2 * CCH + nt * 16 + n] = c[r];
    }
  }
}

// ---------------------------------------------------------------------------
// Small elementwise helpers
// ---------------------------------------------------------------------------
__global__ void k_f2b(bf16* dst, const float* src, long nelem) {
  for (long i = (long)blockIdx.x * 256 + threadIdx.x; i < nelem; i += (long)gridDim.x * 256)
    dst[i] = (bf16)src[i];
}
__global__ void k_zero(float* p, long nelem) {
  for (long i = (long)blockIdx.x * 256 + threadIdx.x; i < nelem; i += (long)gridDim.x * 256)
    p[i] = 0.f;
}
__global__ void k_basis(bf16* basis, const float* dist, int E) {
  const float step = 8.0f / 511.0f;
  const float coeff = -0.5f / (step * step);
  long n = (long)E * 512;
  for (long i = (long)blockIdx.x * 256 + threadIdx.x; i < n; i += (long)gridDim.x * 256) {
    int e = (int)(i >> 9), g = (int)(i & 511);
    float d = dist[e] - (float)g * step;
    basis[i] = (bf16)__expf(coeff * d * d);
  }
}
__global__ void k_combine(bf16* xe0, const float* cxd, const float* semb,
                          const float* temb, const int* an, const int* ei, int E) {
  long n = (long)E * 128;
  for (long i = (long)blockIdx.x * 256 + threadIdx.x; i < n; i += (long)gridDim.x * 256) {
    int e = (int)(i >> 7), c = (int)(i & 127);
    float v = cxd[i] + semb[an[ei[e]] * 128 + c] + temb[an[ei[E + e]] * 128 + c];
    xe0[i] = (bf16)silu_f(v);
  }
}
__global__ void k_perm_build(int* perm, int* inv) {
  if (threadIdx.x == 0 && blockIdx.x == 0) {
    int idx = 0;
    for (int l = 0; l <= 6; ++l) perm[idx++] = l * l + l;
    for (int m = 1; m <= 6; ++m) {
      for (int l = m; l <= 6; ++l) perm[idx++] = l * l + l - m;
      for (int l = m; l <= 6; ++l) perm[idx++] = l * l + l + m;
    }
    for (int j = 0; j < 49; ++j) inv[perm[j]] = j;
  }
}
__global__ void k_tgperm(bf16* tgp, const float* tgm, const int* perm) {
  int i = blockIdx.x * 256 + threadIdx.x;
  if (i < GPTS * NCOEF) {
    int g = i / NCOEF, j = i % NCOEF;
    tgp[i] = (bf16)tgm[g * NCOEF + perm[j]];
  }
}

// ---------------------------------------------------------------------------
// Host launcher
// ---------------------------------------------------------------------------
static inline void gemm(hipStream_t s, const bf16* A, long lda, long acol,
                        const bf16* B, long ldb, int M, int N, int K, int flags,
                        const float* bias, long bcol, const bf16* gate, long gld,
                        long gcol, float* outf, bf16* outb, long ldc, long ccol,
                        float sign) {
  dim3 g(N / 32, M / 64);
  gemm_e_kernel<<<g, 256, 0, s>>>(A, lda, acol, B, ldb, K, flags, bias, bcol,
                                  gate, gld, gcol, outf, outb, ldc, ccol, sign);
}

extern "C" void kernel_launch(void* const* d_in, const int* in_sizes, int n_in,
                              void* d_out, int out_size, void* d_ws, size_t ws_size,
                              hipStream_t stream) {
  const int E = in_sizes[1];                // 8000 (multiple of 64)
  const int NA = in_sizes[0] / RS;          // 800 atoms

  const float* f_x      = (const float*)d_in[0];
  const float* f_dist   = (const float*)d_in[1];
  const float* f_wig    = (const float*)d_in[2];
  const float* f_winv   = (const float*)d_in[3];
  const float* f_tg     = (const float*)d_in[4];
  const float* f_fg     = (const float*)d_in[5];
  const float* f_wd1    = (const float*)d_in[6];
  const float* f_bd1    = (const float*)d_in[7];
  const float* f_semb   = (const float*)d_in[8];
  const float* f_temb   = (const float*)d_in[9];
  const float* f_we1    = (const float*)d_in[10];
  const float* f_be1    = (const float*)d_in[11];
  const float* f_ws1    = (const float*)d_in[12];
  const float* f_bs1    = (const float*)d_in[13];
  const float* f_ws2    = (const float*)d_in[14];
  const float* f_bs2    = (const float*)d_in[15];
  const float* f_ws3    = (const float*)d_in[16];
  const float* f_bs3    = (const float*)d_in[17];
  const int*   i_an     = (const int*)d_in[18];
  const int*   i_ei     = (const int*)d_in[19];
  float* out = (float*)d_out;

  char* ws = (char*)d_ws;
  size_t off = 0;
  auto alloc = [&](size_t bytes) -> char* {
    off = (off + 255) & ~(size_t)255;
    char* p = ws + off;
    off += bytes;
    return p;
  };

  int*  perm   = (int*)alloc(49 * 4);
  int*  invp   = (int*)alloc(49 * 4);
  bf16* wd1b   = (bf16*)alloc((size_t)128 * 512 * 2);
  bf16* we1b   = (bf16*)alloc((size_t)128 * 128 * 2);
  bf16* tgb    = (bf16*)alloc((size_t)GPTS * NCOEF * 2);
  bf16* tgpb   = (bf16*)alloc((size_t)GPTS * NCOEF * 2);
  bf16* fgb    = (bf16*)alloc((size_t)NCOEF * GPTS * 2);
  bf16* ws1b   = (bf16*)alloc((size_t)128 * 256 * 2);
  bf16* ws2b   = (bf16*)alloc((size_t)128 * 128 * 2);
  bf16* ws3b   = (bf16*)alloc((size_t)128 * 128 * 2);
  bf16* wd0b[2], *w1m0b[2], *w2m0b[2], *wdmb[2], *w1rb[2], *w2rb[2], *w1ib[2], *w2ib[2];
  bf16* g0b[2], *gmb[2];
  for (int b = 0; b < 2; ++b) {
    wd0b[b]  = (bf16*)alloc((size_t)256 * 128 * 2);
    w1m0b[b] = (bf16*)alloc((size_t)256 * 896 * 2);
    w2m0b[b] = (bf16*)alloc((size_t)896 * 256 * 2);
    wdmb[b]  = (bf16*)alloc((size_t)3072 * 128 * 2);
    w1rb[b]  = (bf16*)alloc((size_t)6 * 256 * 768 * 2);
    w2rb[b]  = (bf16*)alloc((size_t)6 * 768 * 256 * 2);
    w1ib[b]  = (bf16*)alloc((size_t)6 * 256 * 768 * 2);
    w2ib[b]  = (bf16*)alloc((size_t)6 * 768 * 256 * 2);
    g0b[b]   = (bf16*)alloc((size_t)E * 256 * 2);
    gmb[b]   = (bf16*)alloc((size_t)E * 3072 * 2);
  }
  bf16*  basisb = (bf16*)alloc((size_t)E * 512 * 2);
  float* cxd    = (float*)alloc((size_t)E * 128 * 4);
  bf16*  xe0    = (bf16*)alloc((size_t)E * 128 * 2);
  bf16*  xedge  = (bf16*)alloc((size_t)E * 128 * 2);
  bf16*  xrot   = (bf16*)alloc((size_t)E * RS * 2);
  bf16*  ub[4];
  for (int i = 0; i < 4; ++i) ub[i] = (bf16*)alloc((size_t)E * 256 * 2);
  float* msg  = (float*)alloc((size_t)E * RS * 4);
  float* xmsg = (float*)alloc((size_t)NA * RS * 4);

  // ---- weight conversions (every call; deterministic, cheap, L2-resident) ----
  auto cv = [&](bf16* dst, const float* src, long n) {
    long g = (n + 255) / 256; if (g > 4096) g = 4096;
    k_f2b<<<dim3((unsigned)g), 256, 0, stream>>>(dst, src, n);
  };
  k_perm_build<<<1, 1, 0, stream>>>(perm, invp);
  cv(wd1b, f_wd1, 128 * 512);  cv(we1b, f_we1, 128 * 128);
  cv(tgb, f_tg, GPTS * NCOEF); cv(fgb, f_fg, NCOEF * GPTS);
  cv(ws1b, f_ws1, 128 * 256);  cv(ws2b, f_ws2, 128 * 128); cv(ws3b, f_ws3, 128 * 128);
  k_tgperm<<<(GPTS * NCOEF + 255) / 256, 256, 0, stream>>>(tgpb, f_tg, perm);
  const float* bd0[2];  const float* bdm[2];
  for (int b = 0; b < 2; ++b) {
    int b0 = 20 + b * 10;
    cv(wd0b[b],  (const float*)d_in[b0 + 0], 256 * 128);
    cv(w1m0b[b], (const float*)d_in[b0 + 2], 256 * 896);
    cv(w2m0b[b], (const float*)d_in[b0 + 3], 896 * 256);
    cv(wdmb[b],  (const float*)d_in[b0 + 4], 3072 * 128);
    cv(w1rb[b],  (const float*)d_in[b0 + 6], (long)6 * 256 * 768);
    cv(w2rb[b],  (const float*)d_in[b0 + 7], (long)6 * 768 * 256);
    cv(w1ib[b],  (const float*)d_in[b0 + 8], (long)6 * 256 * 768);
    cv(w2ib[b],  (const float*)d_in[b0 + 9], (long)6 * 768 * 256);
    bd0[b] = (const float*)d_in[b0 + 1];
    bdm[b] = (const float*)d_in[b0 + 5];
  }

  // ---- edge embedding ----
  k_basis<<<4096, 256, 0, stream>>>(basisb, f_dist, E);
  gemm(stream, basisb, 512, 0, wd1b, 512, E, 128, 512, EP_BIAS | EP_F32ST,
       f_bd1, 0, nullptr, 0, 0, cxd, nullptr, 128, 0, 1.f);
  k_combine<<<4096, 256, 0, stream>>>(xe0, cxd, f_semb, f_temb, i_an, i_ei, E);
  gemm(stream, xe0, 128, 0, we1b, 128, E, 128, 128, EP_BIAS | EP_SILU,
       f_be1, 0, nullptr, 0, 0, nullptr, xedge, 128, 0, 1.f);

  // ---- gates ----
  for (int b = 0; b < 2; ++b) {
    gemm(stream, xedge, 128, 0, wd0b[b], 128, E, 256, 128, EP_BIAS | EP_SILU,
         bd0[b], 0, nullptr, 0, 0, nullptr, g0b[b], 256, 0, 1.f);
    gemm(stream, xedge, 128, 0, wdmb[b], 128, E, 3072, 128, EP_BIAS | EP_SILU,
         bdm[b], 0, nullptr, 0, 0, nullptr, gmb[b], 3072, 0, 1.f);
  }

  // ---- SO(2) blocks (src then tgt), accumulating into msg ----
  k_zero<<<4096, 256, 0, stream>>>(msg, (long)E * RS);
  for (int b = 0; b < 2; ++b) {
    rotate_kernel<<<E, 256, 0, stream>>>(f_wig, f_x, i_ei, E, b, invp, xrot);
    // m = 0
    gemm(stream, xrot, RS, 0, w1m0b[b], 896, E, 256, 896, EP_GATE,
         nullptr, 0, g0b[b], 256, 0, nullptr, ub[0], 256, 0, 1.f);
    gemm(stream, ub[0], 256, 0, w2m0b[b], 256, E, 896, 256, EP_F32ACC,
         nullptr, 0, nullptr, 0, 0, msg, nullptr, RS, 0, 1.f);
    // m = 1..6
    int offr = 7;
    for (int m = 1; m <= 6; ++m) {
      int nsz = 7 - m, kk = nsz * 128;
      const bf16* W1[2] = { w1rb[b] + (long)(m - 1) * 256 * 768,
                            w1ib[b] + (long)(m - 1) * 256 * 768 };
      const bf16* W2[2] = { w2rb[b] + (long)(m - 1) * 768 * 256,
                            w2ib[b] + (long)(m - 1) * 768 * 256 };
      for (int w = 0; w < 2; ++w)
        for (int r = 0; r < 2; ++r)
          gemm(stream, xrot, RS, (long)(offr + r * nsz) * 128, W1[w], 768,
               E, 256, kk, EP_GATE, nullptr, 0,
               gmb[b], 3072, (long)(m - 1) * 512 + w * 256,
               nullptr, ub[w * 2 + r], 256, 0, 1.f);
      // row0 = real(r0) - imag(r1); row1 = real(r1) + imag(r0)
      gemm(stream, ub[0], 256, 0, W2[0], 256, E, kk, 256, EP_F32ACC,
           nullptr, 0, nullptr, 0, 0, msg, nullptr, RS, (long)offr * 128, 1.f);
      gemm(stream, ub[3], 256, 0, W2[1], 256, E, kk, 256, EP_F32ACC,
           nullptr, 0, nullptr, 0, 0, msg, nullptr, RS, (long)offr * 128, -1.f);
      gemm(stream, ub[1], 256, 0, W2[0], 256, E, kk, 256, EP_F32ACC,
           nullptr, 0, nullptr, 0, 0, msg, nullptr, RS, (long)(offr + nsz) * 128, 1.f);
      gemm(stream, ub[2], 256, 0, W2[1], 256, E, kk, 256, EP_F32ACC,
           nullptr, 0, nullptr, 0, 0, msg, nullptr, RS, (long)(offr + nsz) * 128, 1.f);
      offr += 2 * nsz;
    }
  }

  // ---- edge grid path + scatter into atoms ----
  k_zero<<<4096, 256, 0, stream>>>(xmsg, (long)NA * RS);
  grid_edge_kernel<<<E, 256, 0, stream>>>(msg, tgpb, fgb, f_winv, i_ei, E, xmsg);

  // ---- node path ----
  node_kernel<<<NA, 256, 0, stream>>>(f_x, xmsg, tgb, fgb, ws1b, f_bs1, ws2b,
                                      f_bs2, ws3b, f_bs3, out);
}
